// GNNWrapper_86938728006236
// MI455X (gfx1250) — matrix operand order repaired and checked by reading, not compiled, not run on hardware
//
#include <hip/hip_runtime.h>
#include <hip/hip_bf16.h>
#include <math.h>

// Graph/model dims (match reference)
#define NN 32     // nodes
#define FF 16     // node feats
#define SS 8      // edge feats
#define CC 128    // ECC channels
#define HH 256    // kernel-net hidden
#define DD 256    // dense out
#define BB 64     // batch
#define OBS_STRIDE 9728

typedef __attribute__((ext_vector_type(16))) __bf16 v16bf;
typedef __attribute__((ext_vector_type(8)))  __bf16 v8bf;
typedef __attribute__((ext_vector_type(8)))  float  v8f;

#define H1LD 264   // padded bf16 row stride for [*,256] tiles (16B multiple)
#define W2LD 40    // padded bf16 row stride for [*,32] k-blocks (16B multiple)

// ---- WMMA fragment loaders (CDNA5 wave32 layouts, CDNA5 ISA) ----
// A (16x32 bf16): lane l<16 holds row l, K {0..7,16..23}; lanes>=16 row l-16, K {8..15,24..31}
__device__ __forceinline__ v16bf frag_a(const __bf16* base, int ld) {
  const int l = threadIdx.x & 31;
  const __bf16* p = base + (l & 15) * ld + ((l >> 4) << 3);
  v8bf lo = *(const v8bf*)p;          // K = khi .. khi+7
  v8bf hi = *(const v8bf*)(p + 16);   // K = 16+khi .. 16+khi+7
  return __builtin_shufflevector(lo, hi, 0,1,2,3,4,5,6,7,8,9,10,11,12,13,14,15);
}
// B (32x16 bf16), stored col-major in LDS as [n][k]: lane l<16 col l K {0..7,16..23}; lanes>=16 K {8..15,24..31}
__device__ __forceinline__ v16bf frag_b(const __bf16* base, int ld) {
  const int l = threadIdx.x & 31;
  const __bf16* p = base + (l & 15) * ld + ((l >> 4) << 3);
  v8bf lo = *(const v8bf*)p;
  v8bf hi = *(const v8bf*)(p + 16);
  return __builtin_shufflevector(lo, hi, 0,1,2,3,4,5,6,7,8,9,10,11,12,13,14,15);
}
__device__ __forceinline__ v8f wmma_bf16(v16bf a, v16bf b, v8f c) {
  return __builtin_amdgcn_wmma_f32_16x16x32_bf16(false, a, false, b, (short)0, c, false, false);
}

// ============================================================================
// Kernel A: one workgroup per (b,i). Computes, fully fused in LDS:
//   h1 = relu(E[b,:,i,:] @ W1 + b1)              [32,256]  (VALU, K=8)
//   h2 = relu(h1 @ W2 + b2)                      [32,256]  (WMMA bf16)
//   Wx[h,c] = sum_f Wk[h, c*16+f] * X[b,i,f]     [256,128] (VALU, bf16 to LDS)
//   M  = h2 @ Wx                                 [32,128]  (WMMA bf16)
//   V[b,j,i,c] = A[b,j,i] * (M[j,c] + bkX[c])
// ============================================================================
__global__ __launch_bounds__(256) void ecc_edge_kernel(
    const float* __restrict__ obs,
    const float* __restrict__ W1, const float* __restrict__ b1,
    const float* __restrict__ W2, const float* __restrict__ b2,
    const float* __restrict__ Wk, const float* __restrict__ bk,
    float* __restrict__ V)
{
  __shared__ __align__(16) __bf16 sh_h1[NN * H1LD];
  __shared__ __align__(16) __bf16 sh_h2[NN * H1LD];
  __shared__ __align__(16) __bf16 sh_w2t[HH * W2LD];
  __shared__ __align__(16) __bf16 sh_wxt[CC * H1LD];
  __shared__ float sh_x[FF];
  __shared__ float sh_bkx[CC];
  __shared__ float sh_adj[NN];

  const int t    = threadIdx.x;
  const int lane = t & 31;
  const int wid  = t >> 5;
  const int b    = blockIdx.x >> 5;
  const int i    = blockIdx.x & 31;
  const float* ob = obs + (size_t)b * OBS_STRIDE;

  // ---- stage 0: X[b,i,:] and adjacency column A[b,:,i] ----
  if (t < FF) sh_x[t] = ob[i * FF + t];
  if (t < NN) sh_adj[t] = ob[NN * FF + t * NN + i];
  __syncthreads();

  // bkX[c] = sum_f bk[c*16+f] * X[f]
  if (t < CC) {
    float s = 0.f;
    #pragma unroll
    for (int f = 0; f < FF; ++f) s += bk[t * FF + f] * sh_x[f];
    sh_bkx[t] = s;
  }

  // ---- stage 1: h1 = relu(E @ W1 + b1); thread: j=lane, h-block=wid*32 ----
  {
    const int j  = t & 31;
    const int hb = (t >> 5) << 5;
    const float* ep = ob + NN * FF + NN * NN + (j * NN + i) * SS;
    float e[SS];
    #pragma unroll
    for (int s = 0; s < SS; ++s) e[s] = ep[s];
    for (int u = 0; u < 32; ++u) {
      const int h = hb + u;
      float acc = b1[h];
      #pragma unroll
      for (int s = 0; s < SS; ++s) acc += e[s] * W1[s * HH + h];
      sh_h1[j * H1LD + h] = (__bf16)fmaxf(acc, 0.f);
    }
  }
  __syncthreads();

  // ---- stage 2: h2 = relu(h1 @ W2 + b2) via WMMA; 2 m-tiles x 16 n-tiles ----
  {
    v8f acc[4] = { {}, {}, {}, {} };
    const int m0 = (wid & 1) << 4;
    const int ng = wid >> 1;            // 0..3, covers n-tiles ng*4 .. ng*4+3
    for (int kb = 0; kb < 8; ++kb) {
      const int k0 = kb << 5;
      // stage W2^T k-block (coalesced f32 reads, bf16 to LDS)
      for (int k = 0; k < 32; ++k)
        sh_w2t[t * W2LD + k] = (__bf16)W2[(k0 + k) * HH + t];
      __syncthreads();
      v16bf a = frag_a(sh_h1 + m0 * H1LD + k0, H1LD);
      #pragma unroll
      for (int c = 0; c < 4; ++c) {
        v16bf bfr = frag_b(sh_w2t + (ng * 4 + c) * 16 * W2LD, W2LD);
        acc[c] = wmma_bf16(a, bfr, acc[c]);
      }
      __syncthreads();
    }
    // epilogue: +b2, relu, bf16 -> sh_h2
    const int rbase = m0 + ((lane >> 4) << 3);
    #pragma unroll
    for (int c = 0; c < 4; ++c) {
      const int col = ((ng * 4 + c) << 4) + (lane & 15);
      const float bias = b2[col];
      #pragma unroll
      for (int e = 0; e < 8; ++e)
        sh_h2[(rbase + e) * H1LD + col] = (__bf16)fmaxf(acc[c][e] + bias, 0.f);
    }
  }
  __syncthreads();

  // ---- stage 3: WxT[c][h] = bf16( sum_f Wk[h][c*16+f] * X[f] ) ----
  {
    float xr[FF];
    #pragma unroll
    for (int f = 0; f < FF; ++f) xr[f] = sh_x[f];
    const int c = t & 127;
    const int hbase = t >> 7;           // 0 or 1
    for (int r = 0; r < 128; ++r) {
      const int h = hbase + (r << 1);
      const float* wp = Wk + (size_t)h * (CC * FF) + c * FF;  // 16 contiguous floats
      float s = 0.f;
      #pragma unroll
      for (int f = 0; f < FF; ++f) s += wp[f] * xr[f];
      sh_wxt[c * H1LD + h] = (__bf16)s;
    }
  }
  __syncthreads();

  // ---- stage 4: M = h2 @ Wx via WMMA; 2 m-tiles x 8 n-tiles; write V ----
  {
    v8f acc0 = {}; v8f acc1 = {};
    const int m0 = (wid & 1) << 4;
    const int ng = wid >> 1;            // n-tiles ng*2, ng*2+1
    #pragma unroll
    for (int kb = 0; kb < 8; ++kb) {
      const int k0 = kb << 5;
      v16bf a  = frag_a(sh_h2 + m0 * H1LD + k0, H1LD);
      v16bf b0 = frag_b(sh_wxt + (ng * 2 + 0) * 16 * H1LD + k0, H1LD);
      v16bf b1f = frag_b(sh_wxt + (ng * 2 + 1) * 16 * H1LD + k0, H1LD);
      acc0 = wmma_bf16(a, b0, acc0);
      acc1 = wmma_bf16(a, b1f, acc1);
    }
    const int rbase = m0 + ((lane >> 4) << 3);
    #pragma unroll
    for (int cs = 0; cs < 2; ++cs) {
      v8f& A = cs ? acc1 : acc0;
      const int cc = ((ng * 2 + cs) << 4) + (lane & 15);
      const float bx = sh_bkx[cc];
      #pragma unroll
      for (int e = 0; e < 8; ++e) {
        const int j = rbase + e;
        V[(((size_t)b * NN + j) * NN + i) * CC + cc] = (A[e] + bx) * sh_adj[j];
      }
    }
  }
}

// ============================================================================
// Kernel B: Xc[b,j,c] = relu( sum_i V[b,j,i,c] + X[b,j,:]@Wroot[:,c] + bconv[c] )
// ============================================================================
__global__ __launch_bounds__(128) void ecc_node_kernel(
    const float* __restrict__ obs, const float* __restrict__ V,
    const float* __restrict__ Wroot, const float* __restrict__ bconv,
    float* __restrict__ Xc)
{
  const int bj = blockIdx.x;        // b*32 + j
  const int b  = bj >> 5, j = bj & 31;
  const int c  = threadIdx.x;       // 0..127
  const float* ob = obs + (size_t)b * OBS_STRIDE;
  float s = bconv[c];
  #pragma unroll
  for (int f = 0; f < FF; ++f) s += ob[j * FF + f] * Wroot[f * CC + c];
  const float* vp = V + (size_t)bj * NN * CC + c;
  float acc = 0.f;
  #pragma unroll
  for (int i = 0; i < NN; ++i) acc += vp[i * CC];
  Xc[(size_t)bj * CC + c] = fmaxf(acc + s, 0.f);
}

// ============================================================================
// Kernel C: global-attn-sum pool (softmax over 32 nodes) + Dense(tanh)
// ============================================================================
__global__ __launch_bounds__(256) void pool_dense_kernel(
    const float* __restrict__ Xc, const float* __restrict__ attn_w,
    const float* __restrict__ Wd, const float* __restrict__ bd,
    float* __restrict__ out)
{
  __shared__ float sx[NN * CC];
  __shared__ float slog[NN];
  __shared__ float sattn[NN];
  __shared__ float spool[CC];
  const int b = blockIdx.x, t = threadIdx.x;
  for (int idx = t; idx < NN * CC; idx += 256)
    sx[idx] = Xc[(size_t)b * NN * CC + idx];
  __syncthreads();
  if (t < NN) {
    float s = 0.f;
    #pragma unroll
    for (int c = 0; c < CC; ++c) s += sx[t * CC + c] * attn_w[c];
    slog[t] = s;
  }
  __syncthreads();
  if (t == 0) {
    float m = slog[0];
    for (int n = 1; n < NN; ++n) m = fmaxf(m, slog[n]);
    float sum = 0.f;
    for (int n = 0; n < NN; ++n) { float e = __expf(slog[n] - m); sattn[n] = e; sum += e; }
    const float inv = 1.f / sum;
    for (int n = 0; n < NN; ++n) sattn[n] *= inv;
  }
  __syncthreads();
  if (t < CC) {
    float p = 0.f;
    #pragma unroll
    for (int n = 0; n < NN; ++n) p += sattn[n] * sx[n * CC + t];
    spool[t] = p;
  }
  __syncthreads();
  {
    float s = bd[t];
    #pragma unroll
    for (int c = 0; c < CC; ++c) s += spool[c] * Wd[c * DD + t];
    out[(size_t)b * DD + t] = tanhf(s);
  }
}

extern "C" void kernel_launch(void* const* d_in, const int* in_sizes, int n_in,
                              void* d_out, int out_size, void* d_ws, size_t ws_size,
                              hipStream_t stream) {
  (void)in_sizes; (void)n_in; (void)out_size; (void)ws_size;
  const float* obs   = (const float*)d_in[0];
  const float* W1    = (const float*)d_in[1];
  const float* b1    = (const float*)d_in[2];
  const float* W2    = (const float*)d_in[3];
  const float* b2    = (const float*)d_in[4];
  const float* Wk    = (const float*)d_in[5];
  const float* bk    = (const float*)d_in[6];
  const float* Wroot = (const float*)d_in[7];
  const float* bconv = (const float*)d_in[8];
  const float* attn  = (const float*)d_in[9];
  const float* Wd    = (const float*)d_in[10];
  const float* bd    = (const float*)d_in[11];
  float* out = (float*)d_out;

  float* V  = (float*)d_ws;                                  // [B,N,N,C] f32 = 33.5 MB
  float* Xc = V + (size_t)BB * NN * NN * CC;                 // [B,N,C]   f32 =  1 MB

  ecc_edge_kernel<<<dim3(BB * NN), dim3(256), 0, stream>>>(obs, W1, b1, W2, b2, Wk, bk, V);
  ecc_node_kernel<<<dim3(BB * NN), dim3(128), 0, stream>>>(obs, V, Wroot, bconv, Xc);
  pool_dense_kernel<<<dim3(BB), dim3(256), 0, stream>>>(Xc, attn, Wd, bd, out);
}
